// xLSTM_25838523253522
// MI455X (gfx1250) — compile-verified, matrix operand order checked by reading, not run on hardware
//
#include <hip/hip_runtime.h>
#include <hip/hip_bf16.h>
#include <math.h>

// ---------------------------------------------------------------------------
// xLSTM forward for MI455X (gfx1250, wave32, WMMA).
// Heavy op (mLSTM S x S decayed attention) uses v_wmma_f32_16x16x32_f16
// (f16 inputs, f32 accumulate) in a flash-style online-renormalized loop,
// with LDS-staged, coalesced, branch-free tile loads.
// ---------------------------------------------------------------------------

#define B_   16
#define S_   600
#define NF_  8
#define D_   40
#define DI_  80
#define NH_  4
#define DHS_ 10
#define DHM_ 20
#define FFU_ 64
#define NEG_ (-1e30f)
#define EPS_ 1e-5f

typedef __attribute__((ext_vector_type(16))) _Float16 v16h;
typedef __attribute__((ext_vector_type(8)))  float    v8f;

union F16Frag { v16h v; _Float16 h[16]; };
union F32Frag { v8f  v; float     f[8];  };

__device__ __forceinline__ float sigf(float x)  { return 1.f / (1.f + expf(-x)); }
__device__ __forceinline__ float siluf(float x) { return x * sigf(x); }
__device__ __forceinline__ float logsigf(float x) {
  return fminf(x, 0.f) - log1pf(expf(-fabsf(x)));
}
// ISA 7.12.2: f16 A-matrix 16x32 (and mirrored B 32x16) K index for
// (lane, vgpr pair vi, half e):
__device__ __forceinline__ int wk16(int lane, int vi, int e) {
  return ((vi & 3) * 2 + e) + ((vi >= 4) ? 16 : 0) + ((lane >= 16) ? 8 : 0);
}

// --------------------------- small fused kernels ---------------------------

__global__ void k_in(const float* __restrict__ x, const float* __restrict__ w,
                     const float* __restrict__ b, float* __restrict__ h, int total) {
  int idx = blockIdx.x * blockDim.x + threadIdx.x;
  if (idx >= total) return;
  int token = idx / D_, d = idx % D_;
  float acc = b[d];
  for (int f = 0; f < NF_; ++f) acc += x[token * NF_ + f] * w[f * D_ + d];
  h[idx] = acc;
}

// per-token: ln(h) then @ up (40 x 160) -> xm(80), z(80)
__global__ __launch_bounds__(160)
void k_ln_up(const float* __restrict__ h, const float* __restrict__ lnw,
             const float* __restrict__ up, float* __restrict__ xm, float* __restrict__ z) {
  int token = blockIdx.x, tid = threadIdx.x;
  __shared__ float sh[D_];
  __shared__ float lnv[D_];
  if (tid < D_) sh[tid] = h[(size_t)token * D_ + tid];
  __syncthreads();
  float mu = 0.f;
  for (int i = 0; i < D_; ++i) mu += sh[i];
  mu *= (1.f / D_);
  float var = 0.f;
  for (int i = 0; i < D_; ++i) { float d = sh[i] - mu; var += d * d; }
  var *= (1.f / D_);
  float rstd = rsqrtf(var + EPS_);
  if (tid < D_) lnv[tid] = (sh[tid] - mu) * rstd * lnw[tid];
  __syncthreads();
  float acc = 0.f;
  for (int i = 0; i < D_; ++i) acc += lnv[i] * up[i * 160 + tid];
  if (tid < DI_) xm[(size_t)token * DI_ + tid] = acc;
  else           z [(size_t)token * DI_ + tid - DI_] = acc;
}

// causal depthwise conv (K=4) + SiLU, C channels
__global__ void k_cconv_silu(const float* __restrict__ x, const float* __restrict__ w,
                             const float* __restrict__ b, float* __restrict__ out,
                             int C, int total) {
  int idx = blockIdx.x * blockDim.x + threadIdx.x;
  if (idx >= total) return;
  int c = idx % C;
  int t = (idx / C) % S_;
  int bb = idx / (C * S_);
  float acc = b[c];
  for (int kk = 0; kk < 4; ++kk) {
    int tt = t - 3 + kk;
    if (tt >= 0) acc += x[((size_t)(bb * S_ + tt)) * C + c] * w[c * 4 + kk];
  }
  out[idx] = siluf(acc);
}

// headwise 4x4 block-diagonal projections q,k (from xc) and v (from xm)
__global__ void k_qkv(const float* __restrict__ xc, const float* __restrict__ xm,
                      const float* __restrict__ qw, const float* __restrict__ kw,
                      const float* __restrict__ vw, float* __restrict__ qo,
                      float* __restrict__ ko, float* __restrict__ vo, int total) {
  int idx = blockIdx.x * blockDim.x + threadIdx.x;
  if (idx >= total) return;
  int token = idx / DI_, j = idx % DI_;
  int p = j >> 2, o = j & 3;
  const float* xcp = xc + (size_t)token * DI_ + p * 4;
  const float* xmp = xm + (size_t)token * DI_ + p * 4;
  float aq = 0.f, ak = 0.f, av = 0.f;
  for (int i = 0; i < 4; ++i) {
    aq += qw[p * 16 + o * 4 + i] * xcp[i];
    ak += kw[p * 16 + o * 4 + i] * xcp[i];
    av += vw[p * 16 + o * 4 + i] * xmp[i];
  }
  qo[idx] = aq; ko[idx] = ak; vo[idx] = av;
}

// ipre/fpre = concat(q,k,v)(240) @ (240x4) + bias, stored [b, n, t]
__global__ void k_gates(const float* __restrict__ q, const float* __restrict__ k,
                        const float* __restrict__ v, const float* __restrict__ igw,
                        const float* __restrict__ igb, const float* __restrict__ fgw,
                        const float* __restrict__ fgb, float* __restrict__ ipre,
                        float* __restrict__ fpre, int total) {
  int idx = blockIdx.x * blockDim.x + threadIdx.x;
  if (idx >= total) return;
  int token = idx / NH_, n = idx % NH_;
  int bb = token / S_, t = token % S_;
  const float* qp = q + (size_t)token * DI_;
  const float* kp = k + (size_t)token * DI_;
  const float* vp = v + (size_t)token * DI_;
  float ai = igb[n], af = fgb[n];
  for (int j = 0; j < DI_; ++j) {
    ai += qp[j] * igw[j * NH_ + n];          af += qp[j] * fgw[j * NH_ + n];
    ai += kp[j] * igw[(DI_ + j) * NH_ + n];  af += kp[j] * fgw[(DI_ + j) * NH_ + n];
    ai += vp[j] * igw[(2 * DI_ + j) * NH_ + n];
    af += vp[j] * fgw[(2 * DI_ + j) * NH_ + n];
  }
  size_t o = (size_t)(bb * NH_ + n) * S_ + t;
  ipre[o] = ai; fpre[o] = af;
}

// cumf[t] = cumsum(log_sigmoid(fpre)) per (b, head)
__global__ void k_cumf(const float* __restrict__ fpre, float* __restrict__ cumf) {
  int bn = blockIdx.x * blockDim.x + threadIdx.x;
  if (bn >= B_ * NH_) return;
  const float* fp = fpre + (size_t)bn * S_;
  float* cp = cumf + (size_t)bn * S_;
  float acc = 0.f;
  for (int t = 0; t < S_; ++t) { acc += logsigf(fp[t]); cp[t] = acc; }
}

// ------------------------ WMMA mLSTM attention cell ------------------------
// One wave per (16-row query tile, head, batch). Flash-style: running row max
// m, running |sum C| accumulator, rescaled f32 WMMA accumulators for C @ V.
// All tiles staged through LDS with coalesced, mask-multiplied (branch-free)
// global loads; zero padding lives in LDS so fragment builds are pure ds ops.
__global__ __launch_bounds__(32)
void k_attn(const float* __restrict__ q, const float* __restrict__ k,
            const float* __restrict__ vv, const float* __restrict__ ipre,
            const float* __restrict__ cumf, float* __restrict__ hout) {
  const int tile = blockIdx.x;
  const int n    = blockIdx.y;
  const int b    = blockIdx.z;
  const int lane = threadIdx.x;
  const int rowbase = tile * 16;
  const int nc = lane & 15;
  const int hi = lane >> 4;

  // Stride 33 keeps both row-gather and column-gather patterns conflict-free.
  __shared__ float qs[16][33];   // Q tile: [row-in-tile][dim 0..31, >=20 zero]
  __shared__ float ks[16][33];   // K tile: [key-in-tile][dim 0..31, >=20 zero]
  __shared__ float vs[32][33];   // V tile: [key 0..31 (>=16 zero)][dim 0..31]
  __shared__ float pb[16][33];   // P tile: [row][key 0..31, >=16 zero]

  const float* qp = q  + (size_t)b * S_ * DI_ + n * DHM_;
  const float* kp = k  + (size_t)b * S_ * DI_ + n * DHM_;
  const float* vp = vv + (size_t)b * S_ * DI_ + n * DHM_;
  const float* ip = ipre + (size_t)(b * NH_ + n) * S_;
  const float* cf = cumf + (size_t)(b * NH_ + n) * S_;

  const int ca   = (lane < DHM_) ? lane : (DHM_ - 1);  // clamped dim address
  const float cm = (lane < DHM_) ? 1.f : 0.f;          // dim mask

  // One-time staging: Q tile (coalesced: iteration = row, lane = dim).
#pragma unroll 4
  for (int it = 0; it < 16; ++it) {
    int srow = rowbase + it;
    int ra = (srow < S_) ? srow : (S_ - 1);
    float mask = (srow < S_) ? cm : 0.f;
    qs[it][lane] = qp[(size_t)ra * DI_ + ca] * mask;
  }
  // Zero the never-rewritten pads: V key-rows 16..31, P key-cols 16..31.
#pragma unroll 4
  for (int it = 0; it < 16; ++it) vs[16 + it][lane] = 0.f;
#pragma unroll
  for (int it = 0; it < 8; ++it) {
    int idx = it * 32 + lane;
    pb[idx >> 4][16 + (idx & 15)] = 0.f;
  }
  __syncthreads();

  // A fragment: Q rows (f16, head dim 20 zero-padded to K=32).
  F16Frag aq;
#pragma unroll
  for (int vi = 0; vi < 8; ++vi)
#pragma unroll
    for (int e = 0; e < 2; ++e)
      aq.h[vi * 2 + e] = (_Float16)qs[nc][wk16(lane, vi, e)];

  F32Frag zf;
#pragma unroll
  for (int r = 0; r < 8; ++r) zf.f[r] = 0.f;
  F32Frag acc0 = zf, acc1 = zf;

  float mrun[8], ssum[8], cfrow[8];
#pragma unroll
  for (int r = 0; r < 8; ++r) {
    mrun[r] = NEG_; ssum[r] = 0.f;
    int row = rowbase + r + hi * 8;
    cfrow[r] = cf[(row < S_) ? row : (S_ - 1)];
  }

  const float sc_qk = rsqrtf((float)DHM_);

  for (int j = 0; j <= tile; ++j) {
    const int colbase = j * 16;
    const int col = colbase + nc;
    const int colc = (col < S_) ? col : (S_ - 1);
    const float ipc = ip[colc];
    const float cfc = cf[colc];
    if (j < tile) __builtin_prefetch(kp + (size_t)(colbase + 16) * DI_, 0, 3);

    __syncthreads();  // previous iteration's fragment reads complete
    // Stage K and V tiles: coalesced rows, clamped addresses, mask multiply.
#pragma unroll 4
    for (int it = 0; it < 16; ++it) {
      int srow = colbase + it;
      int ra = (srow < S_) ? srow : (S_ - 1);
      float mask = (srow < S_) ? cm : 0.f;
      ks[it][lane] = kp[(size_t)ra * DI_ + ca] * mask;
      vs[it][lane] = vp[(size_t)ra * DI_ + ca] * mask;
    }
    __syncthreads();

    // B fragment = K^T: element (kk, ncol) = ks[ncol][kk]
    F16Frag bk;
#pragma unroll
    for (int vi = 0; vi < 8; ++vi)
#pragma unroll
      for (int e = 0; e < 2; ++e)
        bk.h[vi * 2 + e] = (_Float16)ks[nc][wk16(lane, vi, e)];

    F32Frag sco;
    sco.v = __builtin_amdgcn_wmma_f32_16x16x32_f16(false, aq.v, false, bk.v,
                                                   (short)0, zf.v, false, false);

    // Online decayed-softmax statistics; write weighted scores to pb.
#pragma unroll
    for (int r = 0; r < 8; ++r) {
      const int M = r + hi * 8;
      const int row = rowbase + M;
      const bool ok = (row < S_) && (col < S_) && (col <= row);
      float logd = ok ? (cfrow[r] - cfc + ipc) : NEG_;
      float rmax = logd;
      for (int offx = 8; offx; offx >>= 1)
        rmax = fmaxf(rmax, __shfl_xor(rmax, offx, 32));
      float mnew  = fmaxf(mrun[r], rmax);
      float scale = expf(mrun[r] - mnew);
      float p     = ok ? expf(logd - mnew) : 0.f;
      float cvv   = sco.f[r] * sc_qk * p;
      float rsum  = cvv;
      for (int offx = 8; offx; offx >>= 1)
        rsum += __shfl_xor(rsum, offx, 32);
      ssum[r] = ssum[r] * scale + rsum;
      acc0.f[r] *= scale;
      acc1.f[r] *= scale;
      mrun[r] = mnew;
      pb[M][nc] = cvv;
    }
    __syncthreads();

    // P -> A fragment; V -> two B fragments (head dim 20 = 16 + 4).
    F16Frag pa, vb0, vb1;
#pragma unroll
    for (int vi = 0; vi < 8; ++vi)
#pragma unroll
      for (int e = 0; e < 2; ++e) {
        int K = wk16(lane, vi, e);
        pa.h [vi * 2 + e] = (_Float16)pb[nc][K];
        vb0.h[vi * 2 + e] = (_Float16)vs[K][nc];
        vb1.h[vi * 2 + e] = (_Float16)vs[K][16 + nc];
      }

    acc0.v = __builtin_amdgcn_wmma_f32_16x16x32_f16(false, pa.v, false, vb0.v,
                                                    (short)0, acc0.v, false, false);
    acc1.v = __builtin_amdgcn_wmma_f32_16x16x32_f16(false, pa.v, false, vb1.v,
                                                    (short)0, acc1.v, false, false);
  }

#pragma unroll
  for (int r = 0; r < 8; ++r) {
    const int M = r + hi * 8;
    const int row = rowbase + M;
    if (row < S_) {
      float norm = fmaxf(fabsf(ssum[r]), expf(-mrun[r])) + 1e-6f;
      float inv = 1.f / norm;
      hout[(size_t)(b * S_ + row) * DI_ + n * DHM_ + nc] = acc0.f[r] * inv;
      if (nc < DHM_ - 16)
        hout[(size_t)(b * S_ + row) * DI_ + n * DHM_ + 16 + nc] = acc1.f[r] * inv;
    }
  }
}

// per-head LN * mh_ln + skip*xc, * silu(z), @ dn (80x40), residual into h
__global__ __launch_bounds__(80)
void k_mhln_dn(const float* __restrict__ hattn, const float* __restrict__ xc,
               const float* __restrict__ z, const float* __restrict__ mhw,
               const float* __restrict__ skip, const float* __restrict__ dn,
               float* __restrict__ h) {
  int token = blockIdx.x, tid = threadIdx.x;
  __shared__ float hm[DI_];
  int head = tid / DHM_;
  const float* ha = hattn + (size_t)token * DI_;
  float mu = 0.f;
  for (int i = 0; i < DHM_; ++i) mu += ha[head * DHM_ + i];
  mu *= (1.f / DHM_);
  float var = 0.f;
  for (int i = 0; i < DHM_; ++i) { float d = ha[head * DHM_ + i] - mu; var += d * d; }
  var *= (1.f / DHM_);
  float rstd = rsqrtf(var + EPS_);
  float val = (ha[tid] - mu) * rstd * mhw[tid] + skip[tid] * xc[(size_t)token * DI_ + tid];
  val *= siluf(z[(size_t)token * DI_ + tid]);
  hm[tid] = val;
  __syncthreads();
  if (tid < D_) {
    float acc = 0.f;
    for (int i = 0; i < DI_; ++i) acc += hm[i] * dn[i * D_ + tid];
    h[(size_t)token * D_ + tid] += acc;
  }
}

// plain LN(h)*w -> out
__global__ __launch_bounds__(64)
void k_ln(const float* __restrict__ h, const float* __restrict__ w,
          float* __restrict__ out) {
  int token = blockIdx.x, tid = threadIdx.x;
  __shared__ float sh[D_];
  if (tid < D_) sh[tid] = h[(size_t)token * D_ + tid];
  __syncthreads();
  if (tid < D_) {
    float mu = 0.f;
    for (int i = 0; i < D_; ++i) mu += sh[i];
    mu *= (1.f / D_);
    float var = 0.f;
    for (int i = 0; i < D_; ++i) { float d = sh[i] - mu; var += d * d; }
    var *= (1.f / D_);
    out[(size_t)token * D_ + tid] = (sh[tid] - mu) * rsqrtf(var + EPS_) * w[tid];
  }
}

// sLSTM gate pre-activations: gx[token][g][head*10+e], g = {i(xc), f(xc), z(x), o(x)}
__global__ void k_sgates(const float* __restrict__ xc, const float* __restrict__ xln,
                         const float* __restrict__ ig, const float* __restrict__ fg,
                         const float* __restrict__ zg, const float* __restrict__ og,
                         float* __restrict__ gx, int total) {
  int idx = blockIdx.x * blockDim.x + threadIdx.x;
  if (idx >= total) return;
  int token = idx / 160, r = idx % 160;
  int g = r / D_, j = r % D_, hh = j / DHS_, e = j % DHS_;
  const float* w = (g == 0) ? ig : (g == 1) ? fg : (g == 2) ? zg : og;
  const float* src = (g < 2) ? xc : xln;
  const float* wp = w + hh * (DHS_ * DHS_) + e * DHS_;
  const float* sp = src + (size_t)token * D_ + hh * DHS_;
  float acc = 0.f;
  for (int i = 0; i < DHS_; ++i) acc += wp[i] * sp[i];
  gx[idx] = acc;
}

// sLSTM sequential scan: one workgroup per batch row, R cached in LDS.
__global__ __launch_bounds__(160)
void k_sscan(const float* __restrict__ gx, const float* __restrict__ R,
             const float* __restrict__ bias, float* __restrict__ y) {
  int b = blockIdx.x, tid = threadIdx.x;
  __shared__ float Rsh[NH_ * DHS_ * 4 * DHS_];  // 1600
  __shared__ float hs[D_], cs[D_], ns[D_], ms[D_], raw[160];
  for (int i = tid; i < NH_ * DHS_ * 4 * DHS_; i += 160) Rsh[i] = R[i];
  if (tid < D_) { hs[tid] = 0.f; cs[tid] = 0.f; ns[tid] = 0.f; ms[tid] = 0.f; }
  __syncthreads();
  int g = tid / D_, j = tid % D_, hh = j / DHS_, e = j % DHS_;
  for (int t = 0; t < S_; ++t) {
    const float* gp = gx + ((size_t)(b * S_ + t)) * 160;
    float acc = gp[tid] + bias[tid];                 // bias flat (4, NH, DHS)
    const float* rp = Rsh + hh * 400 + g * DHS_ + e; // R[hh][d][g][e], stride d = 40
    const float* hp = hs + hh * DHS_;
    for (int d = 0; d < DHS_; ++d) acc += hp[d] * rp[d * 40];
    raw[tid] = acc;
    __syncthreads();
    if (tid < D_) {
      float ir = raw[tid], fr = raw[D_ + tid], zr = raw[2 * D_ + tid], orr = raw[3 * D_ + tid];
      float lfm = ms[tid] + logsigf(fr);
      float mn = fmaxf(ir, lfm);
      float igt = expf(ir - mn), fgt = expf(lfm - mn);
      float cn = fgt * cs[tid] + igt * tanhf(zr);
      float nn = fgt * ns[tid] + igt;
      float hn = sigf(orr) * (cn / nn);
      cs[tid] = cn; ns[tid] = nn; ms[tid] = mn; hs[tid] = hn;
      y[((size_t)(b * S_ + t)) * D_ + tid] = hn;
    }
    __syncthreads();
  }
}

// per-head LN(y)*gn, residual into h
__global__ void k_sout(const float* __restrict__ y, const float* __restrict__ gn,
                       float* __restrict__ h, int total) {
  int idx = blockIdx.x * blockDim.x + threadIdx.x;
  if (idx >= total) return;
  int token = idx / NH_, hh = idx % NH_;
  const float* yp = y + (size_t)token * D_ + hh * DHS_;
  float mu = 0.f;
  for (int i = 0; i < DHS_; ++i) mu += yp[i];
  mu *= (1.f / DHS_);
  float var = 0.f;
  for (int i = 0; i < DHS_; ++i) { float d = yp[i] - mu; var += d * d; }
  var *= (1.f / DHS_);
  float rstd = rsqrtf(var + EPS_);
  for (int i = 0; i < DHS_; ++i)
    h[(size_t)token * D_ + hh * DHS_ + i] += (yp[i] - mu) * rstd * gn[hh * DHS_ + i];
}

// GeLU-gated FFN: u = x@up(40x128); (gelu(u0)*u1)@dn(64x40); residual into h
__global__ __launch_bounds__(128)
void k_ffn(const float* __restrict__ xln2, const float* __restrict__ up,
           const float* __restrict__ dn, float* __restrict__ h) {
  int token = blockIdx.x, tid = threadIdx.x;
  __shared__ float xl[D_];
  __shared__ float ubuf[2 * FFU_];
  __shared__ float gbuf[FFU_];
  if (tid < D_) xl[tid] = xln2[(size_t)token * D_ + tid];
  __syncthreads();
  float u = 0.f;
  for (int i = 0; i < D_; ++i) u += xl[i] * up[i * (2 * FFU_) + tid];
  ubuf[tid] = u;
  __syncthreads();
  if (tid < FFU_) {
    float a = ubuf[tid];
    float ge = 0.5f * a * (1.f + erff(a * 0.70710678118f));
    gbuf[tid] = ge * ubuf[FFU_ + tid];
  }
  __syncthreads();
  if (tid < D_) {
    float acc = 0.f;
    for (int i = 0; i < FFU_; ++i) acc += gbuf[i] * dn[i * D_ + tid];
    h[(size_t)token * D_ + tid] += acc;
  }
}

// post_ln -> d1(40x30) relu -> d2(30x20) relu -> o(20x1)
__global__ __launch_bounds__(64)
void k_head(const float* __restrict__ h, const float* __restrict__ plw,
            const float* __restrict__ d1w, const float* __restrict__ d1b,
            const float* __restrict__ d2w, const float* __restrict__ d2b,
            const float* __restrict__ ow, const float* __restrict__ ob,
            float* __restrict__ out) {
  int token = blockIdx.x, tid = threadIdx.x;
  __shared__ float sh[D_], lnb[D_], a1[30], a2[20];
  if (tid < D_) sh[tid] = h[(size_t)token * D_ + tid];
  __syncthreads();
  if (tid < D_) {
    float mu = 0.f;
    for (int i = 0; i < D_; ++i) mu += sh[i];
    mu *= (1.f / D_);
    float var = 0.f;
    for (int i = 0; i < D_; ++i) { float d = sh[i] - mu; var += d * d; }
    var *= (1.f / D_);
    lnb[tid] = (sh[tid] - mu) * rsqrtf(var + EPS_) * plw[tid];
  }
  __syncthreads();
  if (tid < 30) {
    float acc = d1b[tid];
    for (int i = 0; i < D_; ++i) acc += lnb[i] * d1w[i * 30 + tid];
    a1[tid] = fmaxf(acc, 0.f);
  }
  __syncthreads();
  if (tid < 20) {
    float acc = d2b[tid];
    for (int i = 0; i < 30; ++i) acc += a1[i] * d2w[i * 20 + tid];
    a2[tid] = fmaxf(acc, 0.f);
  }
  __syncthreads();
  if (tid == 0) {
    float acc = ob[0];
    for (int i = 0; i < 20; ++i) acc += a2[i] * ow[i];
    out[token] = acc;
  }
}

// ------------------------------- host side --------------------------------
// Input ordering assumption: d_in[0] = x, then params pytree flattened
// jax-style (dict keys sorted lexicographically, lists in index order):
//   blocks[0] (mLSTM, 14 leaves sorted: conv_b, conv_w, dn, fg_b, fg_w, ig_b,
//              ig_w, kw, ln, mh_ln, qw, skip, up, vw)            -> 1..14
//   blocks[1] (sLSTM, 13 leaves sorted: R, b, conv_b, conv_w, ff_dn, ff_up,
//              fg, gn, ig, ln1, ln2, og, zg)                     -> 15..27
//   blocks[2..4] mLSTM                                            -> 28..41, 42..55, 56..69
//   d1_b=70, d1_w=71, d2_b=72, d2_w=73, in_b=74, in_w=75, o_b=76, o_w=77, post_ln=78

extern "C" void kernel_launch(void* const* d_in, const int* in_sizes, int n_in,
                              void* d_out, int out_size, void* d_ws, size_t ws_size,
                              hipStream_t stream) {
  (void)in_sizes; (void)n_in; (void)out_size; (void)ws_size;
  const int BS = B_ * S_;
  const float* x = (const float*)d_in[0];
  auto P = [&](int i) { return (const float*)d_in[i]; };

  float* W = (float*)d_ws;
  size_t off = 0;
  auto alloc = [&](size_t nf) { float* p = W + off; off += nf; return p; };
  float* h    = alloc((size_t)BS * D_);
  float* xm   = alloc((size_t)BS * DI_);
  float* zb   = alloc((size_t)BS * DI_);
  float* xc   = alloc((size_t)BS * DI_);
  float* qb   = alloc((size_t)BS * DI_);
  float* kb   = alloc((size_t)BS * DI_);
  float* vb   = alloc((size_t)BS * DI_);
  float* ipre = alloc((size_t)B_ * NH_ * S_);
  float* fpre = alloc((size_t)B_ * NH_ * S_);
  float* cumf = alloc((size_t)B_ * NH_ * S_);
  float* xln  = alloc((size_t)BS * D_);
  float* xcs  = alloc((size_t)BS * D_);
  float* gx   = alloc((size_t)BS * 160);
  // Aliases (lifetimes disjoint): attention output reuses xm; sLSTM scan
  // output reuses vb; ln2 output reuses xln.
  float* hattn = xm;
  float* ybuf  = vb;
  float* xln2  = xln;

  k_in<<<(BS * D_ + 255) / 256, 256, 0, stream>>>(x, P(75), P(74), h, BS * D_);

  const int mbase[4] = {1, 28, 42, 56};
  int mi = 0;
  for (int blk = 0; blk < 5; ++blk) {
    if (blk == 1) {  // sLSTM block
      const int sb = 15;
      k_ln<<<BS, 64, 0, stream>>>(h, P(sb + 9), xln);                       // ln1
      k_cconv_silu<<<(BS * D_ + 255) / 256, 256, 0, stream>>>(
          xln, P(sb + 3), P(sb + 2), xcs, D_, BS * D_);                     // conv_w, conv_b
      k_sgates<<<(BS * 160 + 255) / 256, 256, 0, stream>>>(
          xcs, xln, P(sb + 8), P(sb + 6), P(sb + 12), P(sb + 11), gx, BS * 160); // ig, fg, zg, og
      k_sscan<<<B_, 160, 0, stream>>>(gx, P(sb + 0), P(sb + 1), ybuf);      // R, b
      k_sout<<<(BS * NH_ + 255) / 256, 256, 0, stream>>>(ybuf, P(sb + 7), h, BS * NH_); // gn
      k_ln<<<BS, 64, 0, stream>>>(h, P(sb + 10), xln2);                     // ln2
      k_ffn<<<BS, 128, 0, stream>>>(xln2, P(sb + 5), P(sb + 4), h);         // ff_up, ff_dn
    } else {         // mLSTM block
      const int mb = mbase[mi++];
      k_ln_up<<<BS, 160, 0, stream>>>(h, P(mb + 8), P(mb + 12), xm, zb);    // ln, up
      k_cconv_silu<<<(BS * DI_ + 255) / 256, 256, 0, stream>>>(
          xm, P(mb + 1), P(mb + 0), xc, DI_, BS * DI_);                     // conv_w, conv_b
      k_qkv<<<(BS * DI_ + 255) / 256, 256, 0, stream>>>(
          xc, xm, P(mb + 10), P(mb + 7), P(mb + 13), qb, kb, vb, BS * DI_); // qw, kw, vw
      k_gates<<<(BS * NH_ + 255) / 256, 256, 0, stream>>>(
          qb, kb, vb, P(mb + 6), P(mb + 5), P(mb + 4), P(mb + 3),
          ipre, fpre, BS * NH_);                                            // ig_w/b, fg_w/b
      k_cumf<<<1, 64, 0, stream>>>(fpre, cumf);
      k_attn<<<dim3((S_ + 15) / 16, NH_, B_), 32, 0, stream>>>(
          qb, kb, vb, ipre, cumf, hattn);
      k_mhln_dn<<<BS, 80, 0, stream>>>(hattn, xc, zb, P(mb + 9), P(mb + 11),
                                       P(mb + 2), h);                       // mh_ln, skip, dn
    }
  }
  k_head<<<BS, 64, 0, stream>>>(h, P(78), P(71), P(70), P(73), P(72), P(77),
                                P(76), (float*)d_out);
}
